// SupTreeConLoss_35003983463135
// MI455X (gfx1250) — matrix operand order repaired and checked by reading, not compiled, not run on hardware
//
#include <hip/hip_runtime.h>
#include <hip/hip_fp16.h>

typedef _Float16 v16h __attribute__((ext_vector_type(16)));
typedef _Float16 v8h  __attribute__((ext_vector_type(8)));
typedef float    v8f  __attribute__((ext_vector_type(8)));

#define N_ROWS 8192
#define DIM    256
#define TEMP_INV 2.0f   // 1 / 0.5

// ---------------- kernel 1: row-wise L2 normalize, fp32 -> f16 ----------------
__global__ void __launch_bounds__(256)
norm_rows_f16(const float* __restrict__ in0, const float* __restrict__ in1,
              _Float16* __restrict__ outA, _Float16* __restrict__ outB) {
  const float* in = (blockIdx.y == 0) ? in0 : in1;
  _Float16* out   = (blockIdx.y == 0) ? outA : outB;
  const int row = blockIdx.x;
  const int t   = threadIdx.x;                 // 256 threads == DIM
  float x = in[(size_t)row * DIM + t];
  __shared__ float red[256];
  red[t] = x * x;
  __syncthreads();
  for (int s = 128; s > 0; s >>= 1) {
    if (t < s) red[t] += red[t + s];
    __syncthreads();
  }
  float inv = 1.0f / fmaxf(sqrtf(red[0]), 1e-12f);
  out[(size_t)row * DIM + t] = (_Float16)(x * inv);
}

// ---------------- kernel 2: fused WMMA GEMM + exp + masked row sums ----------------
// Block: 256 threads (8 wave32), computes a 128x128 tile of exp(2*a.b^T).
// Wave w owns rows [16w,16w+16) of the tile, 8 column subtiles of 16.
__global__ void __launch_bounds__(256)
contrastive_gemm(const _Float16* __restrict__ A, const _Float16* __restrict__ B,
                 const int* __restrict__ labels,
                 float* __restrict__ posSum, float* __restrict__ allSum) {
  const int rowBase = blockIdx.y * 128;
  const int colBase = blockIdx.x * 128;
  const int t    = threadIdx.x;
  const int wave = t >> 5;        // 0..7
  const int lane = t & 31;
  const int l16  = lane & 15;
  const int half = lane >> 4;     // 0 or 1

  __shared__ int sLabR[128];
  __shared__ int sLabC[128];
  if (t < 128) sLabR[t] = labels[rowBase + t];
  else         sLabC[t - 128] = labels[colBase + (t - 128)];
  __syncthreads();

  v8f acc[8];
  const v8f vzero = {};
#pragma unroll
  for (int j = 0; j < 8; ++j) acc[j] = vzero;

  // A fragment rows: M = l16 (both lane halves), K = 16*half + s (contiguous).
  const _Float16* arow = A + (size_t)(rowBase + wave * 16 + l16) * DIM;

#pragma unroll
  for (int k = 0; k < DIM; k += 32) {
    union { v16h v; v8h h[2]; } af;
    af.h[0] = *(const v8h*)(arow + k + 16 * half);       // K = 16h .. 16h+7
    af.h[1] = *(const v8h*)(arow + k + 16 * half + 8);   // K = 16h+8 .. 16h+15
#pragma unroll
    for (int j = 0; j < 8; ++j) {
      // B fragment: N = l16 of subtile j, K = 16*half..16*half+15 contiguous
      const _Float16* brow = B + (size_t)(colBase + j * 16 + l16) * DIM + k + 16 * half;
      v16h bf = *(const v16h*)brow;
      acc[j] = __builtin_amdgcn_wmma_f32_16x16x32_f16(
          false, af.v, false, bf, (short)0, acc[j], false, false);
    }
  }

  // ---- epilogue: exp(2x), mask, reduce over columns ----
  int labn[8];
#pragma unroll
  for (int j = 0; j < 8; ++j) labn[j] = sLabC[j * 16 + l16];

#pragma unroll
  for (int r = 0; r < 8; ++r) {
    const int mrow = wave * 16 + r + 8 * half;   // C/D layout: VGPR r -> M = r + 8*half
    const int lm = sLabR[mrow];
    float allv = 0.f, posv = 0.f;
#pragma unroll
    for (int j = 0; j < 8; ++j) {
      float e = __expf(TEMP_INV * acc[j][r]);
      allv += e;
      posv += (lm == labn[j]) ? e : 0.f;
    }
    // reduce across the 16 lanes holding N=0..15 (xor<16 stays inside each half-group)
#pragma unroll
    for (int off = 1; off < 16; off <<= 1) {
      allv += __shfl_xor(allv, off, 32);
      posv += __shfl_xor(posv, off, 32);
    }
    if (l16 == 0) {
      atomicAdd(&allSum[rowBase + mrow], allv);
      atomicAdd(&posSum[rowBase + mrow], posv);
    }
  }
}

// ---------------- kernel 3: loss = -mean(log(pos/all)) ----------------
__global__ void __launch_bounds__(256)
finalize_loss(const float* __restrict__ posSum, const float* __restrict__ allSum,
              float* __restrict__ out) {
  const int t = threadIdx.x;
  float s = 0.f;
  for (int i = t; i < N_ROWS; i += 256)
    s += logf(posSum[i] / allSum[i]);
  __shared__ float red[256];
  red[t] = s;
  __syncthreads();
  for (int w = 128; w > 0; w >>= 1) {
    if (t < w) red[t] += red[t + w];
    __syncthreads();
  }
  if (t == 0) out[0] = -red[0] / (float)N_ROWS;
}

extern "C" void kernel_launch(void* const* d_in, const int* in_sizes, int n_in,
                              void* d_out, int out_size, void* d_ws, size_t ws_size,
                              hipStream_t stream) {
  const float* out0  = (const float*)d_in[0];
  const float* out1  = (const float*)d_in[1];
  const int* labels  = (const int*)d_in[2];

  char* ws = (char*)d_ws;
  _Float16* aH = (_Float16*)ws;                                  // 8192*256 f16
  _Float16* bH = aH + (size_t)N_ROWS * DIM;                      // 8192*256 f16
  float* posSum = (float*)(ws + 2ull * N_ROWS * DIM * sizeof(_Float16));
  float* allSum = posSum + N_ROWS;

  hipMemsetAsync(posSum, 0, 2ull * N_ROWS * sizeof(float), stream);
  norm_rows_f16<<<dim3(N_ROWS, 2), 256, 0, stream>>>(out0, out1, aH, bH);
  contrastive_gemm<<<dim3(N_ROWS / 128, N_ROWS / 128), 256, 0, stream>>>(
      aH, bH, labels, posSum, allSum);
  finalize_loss<<<1, 256, 0, stream>>>(posSum, allSum, (float*)d_out);
}
